// LexiconAdapter_33938831573324
// MI455X (gfx1250) — compile-verified, matrix operand order checked by reading, not run on hardware
//
#include <hip/hip_runtime.h>
#include <hip/hip_bf16.h>
#include <stdint.h>

// ---------------------------------------------------------------------------
// LexiconAdapter fused pipeline for MI455X (gfx1250, wave32, WMMA)
//   GEMM1: h  = tanh(emb[40960,200] @ W1[200,768] + b1)      -> bf16 ws
//   GEMM2: wo = h[40960,768] @ W2[768,768] + b2              -> f32 ws
//   GEMM3: q  = layer_output[8192,768] @ attn_W[768,768]     -> f32 ws
//   EPILOG: bilinear dots, masked softmax over W=5, weighted sum,
//           residual add, LayerNorm                          -> d_out
// bf16 WMMA, ping-pong LDS, batched b128 load clauses, packed converts via
// __builtin_convertvector, pointer-advance addressing (no muls in K loop).
// ---------------------------------------------------------------------------

typedef __attribute__((ext_vector_type(16))) __bf16          v16bf;
typedef __attribute__((ext_vector_type(4)))  __bf16          bf16x4;
typedef __attribute__((ext_vector_type(2)))  __bf16          bf16x2;
typedef __attribute__((ext_vector_type(8)))  float           v8f;
typedef __attribute__((ext_vector_type(8)))  unsigned short  ushort8;
typedef __attribute__((ext_vector_type(4)))  float           f32x4;
typedef __attribute__((ext_vector_type(2)))  float           f32x2;
typedef __attribute__((ext_vector_type(4)))  unsigned int    u32x4;

union BF16Frag { ushort8 h[2]; v16bf v; };

__device__ __forceinline__ unsigned short f32_to_bf16_raw(float f) {
    union { __bf16 b; unsigned short s; } c;
    c.b = (__bf16)f;
    return c.s;
}

// vector fptrunc -> v_cvt_pk_bf16_f32 pairs
__device__ __forceinline__ unsigned long long pack4_bf16(f32x4 f) {
    union { bf16x4 v; unsigned long long u; } r;
    r.v = __builtin_convertvector(f, bf16x4);
    return r.u;
}

__device__ __forceinline__ unsigned pack2_bf16(float lo, float hi) {
    f32x2 f = {lo, hi};
    union { bf16x2 v; unsigned u; } p;
    p.v = __builtin_convertvector(f, bf16x2);
    return p.u;
}

__device__ __forceinline__ float fast_tanh(float x) {
    // tanh(x) = 1 - 2/(exp(2x)+1); saturates correctly at +/-inf
    float e = __expf(2.0f * x);
    return 1.0f - 2.0f / (e + 1.0f);
}

#define BM 128
#define BN 64
#define BK 32

// One K-slab of WMMA work: 8 waves, each wave 32x32 = 2x2 tiles of 16x16.
// Fragment layout per ISA: lane%16 = row(A)/col(B), lane/16 selects K halves
// {0..7,16..23} vs {8..15,24..31}; LDS tiles are [m][k] and [n][k] bf16.
__device__ __forceinline__ void wmma_tile_compute(
    const unsigned short* __restrict__ ldsA,
    const unsigned short* __restrict__ ldsB,
    int lane, int wm, int wn, v8f acc[2][2])
{
    const int kb = (lane >> 4) << 3;   // 0 or 8
    BF16Frag af[2], bf[2];
    #pragma unroll
    for (int mt = 0; mt < 2; ++mt) {
        int row = wm * 32 + mt * 16 + (lane & 15);
        const ushort8* p = (const ushort8*)&ldsA[row * BK + kb];
        af[mt].h[0] = p[0];
        af[mt].h[1] = p[2];            // +32 bytes = K halves +16
    }
    #pragma unroll
    for (int nt = 0; nt < 2; ++nt) {
        int col = wn * 32 + nt * 16 + (lane & 15);
        const ushort8* p = (const ushort8*)&ldsB[col * BK + kb];
        bf[nt].h[0] = p[0];
        bf[nt].h[1] = p[2];
    }
    #pragma unroll
    for (int mt = 0; mt < 2; ++mt)
        #pragma unroll
        for (int nt = 0; nt < 2; ++nt)
            acc[mt][nt] = __builtin_amdgcn_wmma_f32_16x16x32_bf16(
                false, af[mt].v, false, bf[nt].v,
                (short)0, acc[mt][nt], false, false);
}

// EPI: 0 = tanh(acc+bias) -> bf16 out ; 1 = acc+bias -> f32 ; 2 = acc -> f32
// ABF16: A operand in global memory is bf16 (true) or f32 (false)
// K: compile-time reduction dim (200 or 768)
template <int EPI, bool ABF16, int K>
__global__ __launch_bounds__(256) void gemm_wmma_kernel(
    const void*  __restrict__ Aptr,   // [M,K] row-major (f32 or bf16)
    const float* __restrict__ Bg,     // [K,N] row-major f32
    const float* __restrict__ bias,   // [N] or nullptr
    void*        __restrict__ Optr,   // [M,N] (bf16 for EPI==0, else f32)
    int M, int N)
{
    constexpr int KTILES = K / BK;            // full tiles
    constexpr int KTAIL  = K - KTILES * BK;   // leftover K columns (0 or 8)
    constexpr int AIT    = ABF16 ? 2 : 4;     // A staging loads per thread
    constexpr int AV     = ABF16 ? 8 : 4;     // elements per A load
    constexpr int AES    = ABF16 ? 2 : 4;     // A element bytes
    static_assert(KTAIL % 4 == 0, "tail must be float4-aligned");

    // Ping-pong double buffers: 2 * (8 KB + 4 KB) = 24 KB LDS
    __shared__ unsigned short ldsA[2][BM * BK];
    __shared__ unsigned short ldsB[2][BN * BK];

    const int tid  = threadIdx.x;
    const int lane = tid & 31;
    const int wave = tid >> 5;
    const int wm   = wave >> 1;     // 0..3
    const int wn   = wave & 1;      // 0..1

    const int m0 = blockIdx.y * BM;
    const int n0 = blockIdx.x * BN;

    v8f acc[2][2] = {};

    // ---- per-thread staging state: precomputed row pointers, advanced by
    //      compile-time strides each K-tile (no muls in the loop) ----
    const char* aP[AIT];
    int aE[AIT], aC[AIT];
    #pragma unroll
    for (int it = 0; it < AIT; ++it) {
        int e = (it * 256 + tid) * AV;   // element index in 128x32 tile
        int r = e >> 5, c = e & 31;
        aE[it] = e;
        aC[it] = c;
        aP[it] = (const char*)Aptr + ((size_t)(m0 + r) * K + c) * AES;
    }
    const int kp = (tid >> 4) << 1;      // even k in 0..30 (K-pair per thread)
    const int nB = (tid & 15) * 4;       // 4 consecutive n per thread
    const size_t bRow = (size_t)N * 4;   // B row stride (bytes)
    const char* bP = (const char*)Bg + (size_t)kp * bRow + ((size_t)(n0 + nB)) * 4;

    // stage: issue ALL global loads first (single clause), then convert/store
    auto stage = [&](unsigned short* dA, unsigned short* dB, bool tail) {
        u32x4 vah[AIT];
        f32x4 vaf[AIT];
        if (ABF16) {
            #pragma unroll
            for (int it = 0; it < AIT; ++it)
                vah[it] = *(const u32x4*)aP[it];
        } else {
            #pragma unroll
            for (int it = 0; it < AIT; ++it) {
                if (!tail || aC[it] < KTAIL)      // 4-aligned c, full vectors
                    vaf[it] = *(const f32x4*)aP[it];
                else
                    vaf[it] = (f32x4){0.f, 0.f, 0.f, 0.f};
            }
        }
        f32x4 b0 = {0.f, 0.f, 0.f, 0.f}, b1 = {0.f, 0.f, 0.f, 0.f};
        if (!tail || kp < KTAIL) {                // kp even, kp+1 < KTAIL too
            b0 = *(const f32x4*)bP;
            b1 = *(const f32x4*)(bP + bRow);
        }
        // ---- convert + LDS store phase ----
        if (ABF16) {
            #pragma unroll
            for (int it = 0; it < AIT; ++it)
                *(u32x4*)&dA[aE[it]] = vah[it];
        } else {
            #pragma unroll
            for (int it = 0; it < AIT; ++it)
                *(unsigned long long*)&dA[aE[it]] = pack4_bf16(vaf[it]);
        }
        #pragma unroll
        for (int j = 0; j < 4; ++j)               // (k,k+1) pair per dword
            *(unsigned*)&dB[(nB + j) * BK + kp] = pack2_bf16(b0[j], b1[j]);
    };
    auto advance = [&]() {
        #pragma unroll
        for (int it = 0; it < AIT; ++it) aP[it] += BK * AES;
        bP += (size_t)BK * bRow;
    };

    // ---- software pipeline: prologue stage, then compute/stage overlap ----
    stage(ldsA[0], ldsB[0], false);
    advance();
    __syncthreads();

    for (int kt = 0; kt < KTILES; ++kt) {
        const int cur = kt & 1;
        // Compute first: WMMAs issue onto the matrix pipe, then the wave
        // stalls on next-tile global loads while they drain.
        wmma_tile_compute(ldsA[cur], ldsB[cur], lane, wm, wn, acc);
        if (kt + 1 < KTILES) {
            stage(ldsA[cur ^ 1], ldsB[cur ^ 1], false);
            advance();
        } else if constexpr (KTAIL > 0) {
            stage(ldsA[cur ^ 1], ldsB[cur ^ 1], true);
        }
        __syncthreads();
    }

    if constexpr (KTAIL > 0) {
        const int cur = KTILES & 1;
        wmma_tile_compute(ldsA[cur], ldsB[cur], lane, wm, wn, acc);
    }

    // ---- epilogue: C/D layout lane%16 = N, (lane/16)*8 + r = M ----
    const int rowBase = m0 + wm * 32;
    const int colBase = n0 + wn * 32;
    #pragma unroll
    for (int mt = 0; mt < 2; ++mt) {
        #pragma unroll
        for (int nt = 0; nt < 2; ++nt) {
            int n     = colBase + nt * 16 + (lane & 15);
            int rbase = rowBase + mt * 16 + ((lane >> 4) << 3);
            float bv  = (EPI == 2) ? 0.0f : bias[n];
            #pragma unroll
            for (int r = 0; r < 8; ++r) {
                int   m = rbase + r;
                float v = acc[mt][nt][r] + bv;
                if (EPI == 0) {
                    ((unsigned short*)Optr)[(size_t)m * N + n] =
                        f32_to_bf16_raw(fast_tanh(v));
                } else {
                    ((float*)Optr)[(size_t)m * N + n] = v;
                }
            }
        }
    }
}

// ---------------------------------------------------------------------------
// Per-token bilinear attention + softmax + residual + LayerNorm.
// One 256-thread block per token; 3 hidden elements per thread (768 = 3*256).
// Reductions: wave32 shfl_xor butterfly + 8-slot LDS cross-wave combine.
// ---------------------------------------------------------------------------
#define HID 768
#define NW  5

__device__ __forceinline__ float block_reduce_sum(float v, float* red8, int tid) {
    #pragma unroll
    for (int m = 16; m > 0; m >>= 1) v += __shfl_xor(v, m, 32);
    __syncthreads();                      // protect red8 from previous use
    if ((tid & 31) == 0) red8[tid >> 5] = v;
    __syncthreads();
    float s = 0.f;
    #pragma unroll
    for (int w = 0; w < 8; ++w) s += red8[w];
    return s;
}

__global__ __launch_bounds__(256) void attn_ln_kernel(
    const float* __restrict__ layer_output, // [T,768]
    const float* __restrict__ q,            // [T,768]
    const float* __restrict__ wo,           // [T*5,768]
    const int*   __restrict__ mask,         // [T*5]
    const float* __restrict__ gamma,
    const float* __restrict__ beta,
    float*       __restrict__ out)          // [T,768]
{
    const int token = blockIdx.x;
    const int tid   = threadIdx.x;

    __shared__ float red8[8];

    const float* lo  = layer_output + (size_t)token * HID;
    const float* qp  = q            + (size_t)token * HID;
    const float* wop = wo           + (size_t)token * NW * HID;

    float qv[3], lov[3], wv[NW][3];
    #pragma unroll
    for (int j = 0; j < 3; ++j) {
        int h  = tid + j * 256;
        qv[j]  = qp[h];
        lov[j] = lo[h];
        #pragma unroll
        for (int w = 0; w < NW; ++w) wv[w][j] = wop[w * HID + h];
    }

    // ---- alpha[w] = <q, wo[w]> ----
    float a[NW];
    #pragma unroll
    for (int w = 0; w < NW; ++w)
        a[w] = block_reduce_sum(qv[0]*wv[w][0] + qv[1]*wv[w][1] + qv[2]*wv[w][2],
                                red8, tid);

    // ---- mask + softmax over W=5 (identical in every thread) ----
    float mx = -1e30f;
    #pragma unroll
    for (int w = 0; w < NW; ++w) {
        float mw = (float)mask[(size_t)token * NW + w];
        a[w] += (1.0f - mw) * -10000.0f;
        mx = fmaxf(mx, a[w]);
    }
    float den = 0.0f;
    #pragma unroll
    for (int w = 0; w < NW; ++w) { a[w] = __expf(a[w] - mx); den += a[w]; }
    float inv = 1.0f / den;
    #pragma unroll
    for (int w = 0; w < NW; ++w) a[w] *= inv;

    // ---- weighted sum + residual ----
    float o[3];
    #pragma unroll
    for (int j = 0; j < 3; ++j) {
        float s = 0.0f;
        #pragma unroll
        for (int w = 0; w < NW; ++w) s += a[w] * wv[w][j];
        o[j] = lov[j] + s;
    }

    // ---- LayerNorm over hidden ----
    float mu = block_reduce_sum(o[0] + o[1] + o[2], red8, tid) * (1.0f / HID);
    float d0 = o[0]-mu, d1 = o[1]-mu, d2 = o[2]-mu;
    float var = block_reduce_sum(d0*d0 + d1*d1 + d2*d2, red8, tid) * (1.0f / HID);
    float rstd = rsqrtf(var + 1e-12f);

    #pragma unroll
    for (int j = 0; j < 3; ++j) {
        int h = tid + j * 256;
        out[(size_t)token * HID + h] = (o[j] - mu) * rstd * gamma[h] + beta[h];
    }
}

// ---------------------------------------------------------------------------
extern "C" void kernel_launch(void* const* d_in, const int* in_sizes, int n_in,
                              void* d_out, int out_size, void* d_ws, size_t ws_size,
                              hipStream_t stream) {
    (void)in_sizes; (void)n_in; (void)out_size; (void)ws_size;

    const float* layer_output = (const float*)d_in[0];   // [16,512,768]
    const float* emb          = (const float*)d_in[1];   // [16,512,5,200]
    const int*   mask         = (const int*)  d_in[2];   // [16,512,5]
    const float* W1           = (const float*)d_in[3];   // [200,768]
    const float* b1           = (const float*)d_in[4];   // [768]
    const float* W2           = (const float*)d_in[5];   // [768,768]
    const float* b2           = (const float*)d_in[6];   // [768]
    const float* attn_W       = (const float*)d_in[7];   // [768,768]
    const float* ln_gamma     = (const float*)d_in[8];   // [768]
    const float* ln_beta      = (const float*)d_in[9];   // [768]
    float*       out          = (float*)d_out;           // [16,512,768]

    const int T  = 16 * 512;       // 8192 tokens
    const int MW = T * 5;          // 40960 token-word rows
    const int N  = 768;

    // Workspace layout (~214 MB):
    char* ws = (char*)d_ws;
    unsigned short* h_bf16 = (unsigned short*)ws;                        // 40960*768*2
    float* wo = (float*)(ws + (size_t)MW * N * sizeof(unsigned short));  // 40960*768*4
    float* q  = (float*)((char*)wo + (size_t)MW * N * sizeof(float));    // 8192*768*4

    dim3 blk(256);

    // GEMM1: h = tanh(emb @ W1 + b1), f32 A, bf16 out, K=200
    gemm_wmma_kernel<0, false, 200><<<dim3(N / BN, MW / BM), blk, 0, stream>>>(
        emb, W1, b1, h_bf16, MW, N);

    // GEMM2: wo = h @ W2 + b2, bf16 A, f32 out, K=768
    gemm_wmma_kernel<1, true, 768><<<dim3(N / BN, MW / BM), blk, 0, stream>>>(
        h_bf16, W2, b2, wo, MW, N);

    // GEMM3: q = layer_output @ attn_W, f32 A, f32 out, K=768
    gemm_wmma_kernel<2, false, 768><<<dim3(N / BN, T / BM), blk, 0, stream>>>(
        layer_output, attn_W, nullptr, q, T, N);

    // Attention + softmax + residual + LayerNorm
    attn_ln_kernel<<<dim3(T), blk, 0, stream>>>(
        layer_output, q, wo, mask, ln_gamma, ln_beta, out);
}
